// SymbolQueryDecoder_34007551050473
// MI455X (gfx1250) — compile-verified, matrix operand order checked by reading, not run on hardware
//
#include <hip/hip_runtime.h>

// SymbolQueryDecoder: B=1, N=32, T=128, D=256, H=8 (hd=32), L=N*T=4096
// bf16 WMMA (v_wmma_f32_16x16x32_bf16) for all GEMMs + flash attention,
// fp32 LayerNorm / softmax / residual. Working set ~12MB -> L2-resident.

#define LN_EPS 1e-5f
constexpr int Nsym = 32;
constexpr int Tt   = 128;
constexpr int D    = 256;
constexpr int Hh   = 8;
constexpr int HD   = 32;
constexpr int L    = Nsym * Tt; // 4096

typedef unsigned short ushort_t;
typedef __attribute__((ext_vector_type(16))) __bf16 v16bf;
typedef __attribute__((ext_vector_type(8)))  float  v8f;

struct U8 { uint4 lo, hi; };

__device__ __forceinline__ ushort_t f2bf(float f) {
  unsigned x = __builtin_bit_cast(unsigned, f);
  unsigned r = x + 0x7FFFu + ((x >> 16) & 1u); // round-to-nearest-even
  return (ushort_t)(r >> 16);
}

// Fragment loader: 16-bit A/B fragment for 16x16x32 WMMA.
// Per lane: kb = (lane/16)*8; elements 0-7 from pa+kb, elements 8-15 from pb+kb.
__device__ __forceinline__ v16bf load_frag2(const ushort_t* pa, const ushort_t* pb, int lane) {
  const int kb = (lane >> 4) << 3;
  U8 u;
  u.lo = *(const uint4*)(pa + kb);
  u.hi = *(const uint4*)(pb + kb);
  return __builtin_bit_cast(v16bf, u);
}
__device__ __forceinline__ v16bf load_frag(const ushort_t* prow, int k0, int lane) {
  return load_frag2(prow + k0, prow + k0 + 16, lane);
}

// One wave computes a 16x16 fp32 tile of A[M,K] * W[N,K]^T (both row-major bf16).
__device__ __forceinline__ v8f wave_gemm(const ushort_t* A, int lda, const ushort_t* W, int ldw,
                                         int row0, int col0, int K, int lane) {
  v8f acc = {};
  const ushort_t* ar = A + (size_t)(row0 + (lane & 15)) * lda;
  const ushort_t* wr = W + (size_t)(col0 + (lane & 15)) * ldw;
  for (int k0 = 0; k0 < K; k0 += 32) {
    v16bf a = load_frag(ar, k0, lane);
    v16bf b = load_frag(wr, k0, lane);
    acc = __builtin_amdgcn_wmma_f32_16x16x32_bf16(false, a, false, b, (short)0, acc, false, false);
  }
  return acc;
}

// ---------------- kernels ----------------

__global__ void k_convert_bf16(const float* __restrict__ in, ushort_t* __restrict__ out, int n) {
  int i = blockIdx.x * 256 + threadIdx.x;
  if (i < n) out[i] = f2bf(in[i]);
}

// LayerNorm over D=256 (one row per block) -> x_bf; also h -> kv_bf.
__global__ __launch_bounds__(256) void k_ln(const float* __restrict__ h,
                                            const float* __restrict__ g,
                                            const float* __restrict__ b,
                                            ushort_t* __restrict__ x_bf,
                                            ushort_t* __restrict__ kv_bf) {
  const int row = blockIdx.x;
  const int tid = threadIdx.x;
  float v = h[(size_t)row * D + tid];
  float s = v, q = v * v;
#pragma unroll
  for (int off = 1; off < 32; off <<= 1) {
    s += __shfl_xor(s, off, 32);
    q += __shfl_xor(q, off, 32);
  }
  __shared__ float ss[8], sq[8];
  if ((tid & 31) == 0) { ss[tid >> 5] = s; sq[tid >> 5] = q; }
  __syncthreads();
  float S = 0.f, Q2 = 0.f;
#pragma unroll
  for (int w = 0; w < 8; ++w) { S += ss[w]; Q2 += sq[w]; }
  float mean = S * (1.f / D);
  float var  = Q2 * (1.f / D) - mean * mean;
  float rstd = rsqrtf(var + LN_EPS);
  float x = (v - mean) * rstd * g[tid] + b[tid];
  x_bf[(size_t)row * D + tid]  = f2bf(x);
  kv_bf[(size_t)row * D + tid] = f2bf(v);
}

// q = x @ wq^T + bq + sym_emb[row/T]  -> bf16
__global__ __launch_bounds__(256) void k_qproj(const ushort_t* __restrict__ x_bf,
                                               const ushort_t* __restrict__ wq_bf,
                                               const float* __restrict__ bq,
                                               const float* __restrict__ sym,
                                               ushort_t* __restrict__ qf_bf) {
  const int lane = threadIdx.x & 31;
  const int wave = (blockIdx.x << 3) + (threadIdx.x >> 5);
  const int mt = wave >> 4, nt = wave & 15;
  v8f acc = wave_gemm(x_bf, D, wq_bf, D, mt * 16, nt * 16, D, lane);
  const int col   = nt * 16 + (lane & 15);
  const int rbase = mt * 16 + ((lane >> 4) << 3);
#pragma unroll
  for (int g2 = 0; g2 < 8; ++g2) {
    int gr = rbase + g2;
    int n  = gr >> 7; // /T
    float v = acc[g2] + bq[col] + sym[n * D + col];
    qf_bf[(size_t)gr * D + col] = f2bf(v);
  }
}

// Q/K/V projections. blockIdx.y: 0=Q (scaled 1/sqrt(hd), from qf), 1=K, 2=V (stored transposed).
__global__ __launch_bounds__(256) void k_qkv(const ushort_t* __restrict__ qf_bf,
                                             const ushort_t* __restrict__ kv_bf,
                                             const ushort_t* __restrict__ inw_bf,
                                             const float* __restrict__ in_b,
                                             ushort_t* __restrict__ Qh,
                                             ushort_t* __restrict__ Kh,
                                             ushort_t* __restrict__ Vt) {
  const int which = blockIdx.y;
  const int lane = threadIdx.x & 31;
  const int wave = (blockIdx.x << 3) + (threadIdx.x >> 5);
  const int mt = wave >> 4, nt = wave & 15;
  const ushort_t* A = (which == 0) ? qf_bf : kv_bf;
  const ushort_t* W = inw_bf + (size_t)which * D * D;
  v8f acc = wave_gemm(A, D, W, D, mt * 16, nt * 16, D, lane);
  const int col   = nt * 16 + (lane & 15);
  const float bias  = in_b[which * D + col];
  const float scale = (which == 0) ? 0.17677669529663687f : 1.f; // 1/sqrt(32)
  const int hh = col >> 5, d = col & 31;
  const int rbase = mt * 16 + ((lane >> 4) << 3);
#pragma unroll
  for (int g2 = 0; g2 < 8; ++g2) {
    int gr = rbase + g2;
    ushort_t v = f2bf((acc[g2] + bias) * scale);
    if (which == 0)      Qh[((size_t)hh * L + gr) * HD + d] = v;
    else if (which == 1) Kh[((size_t)hh * L + gr) * HD + d] = v;
    else                 Vt[((size_t)hh * HD + d) * L + gr] = v;
  }
}

// Flash attention: one wave per (head, 16-query tile); online softmax; 32 keys/iter.
// Causal over time only: visit key time-tiles tk <= tq_tile; mask diagonal tile elementwise.
__global__ __launch_bounds__(256) void k_attn(const ushort_t* __restrict__ Qh,
                                              const ushort_t* __restrict__ Kh,
                                              const ushort_t* __restrict__ Vt,
                                              ushort_t* __restrict__ ctx) {
  __shared__ alignas(16) ushort_t Pl[8][16][32];
  const int lane = threadIdx.x & 31;
  const int w    = threadIdx.x >> 5;
  const int wave = (blockIdx.x << 3) + w;
  const int head  = wave >> 8;   // 0..7
  const int qtile = wave & 255;  // 0..255
  const int tqt   = qtile & 7;   // query time-tile within symbol (T/16=8)
  const int q0    = qtile * 16;
  const ushort_t* Qp = Qh + (size_t)head * L * HD;
  const ushort_t* Kp = Kh + (size_t)head * L * HD;
  const ushort_t* Vp = Vt + (size_t)head * HD * L;

  const int rl = ((lane >> 4) << 3); // this lane's accum rows: rl..rl+7
  const int cl = lane & 15;          // this lane's accum column

  v16bf qf = load_frag(Qp + (size_t)(q0 + cl) * HD, 0, lane);

  float m[8], ssum[8];
  v8f O0 = {}, O1 = {};
#pragma unroll
  for (int g2 = 0; g2 < 8; ++g2) { m[g2] = -1e30f; ssum[g2] = 0.f; }

  const int per = tqt + 1;      // key time-tiles visited per symbol
  const int NT  = Nsym * per;   // total 16-key tiles (always even)

  for (int i = 0; i < NT; i += 2) {
    const int nsA = i / per,       tkA = i - nsA * per;
    const int jB  = i + 1;
    const int nsB = jB / per,      tkB = jB - nsB * per;
    const int k0a = nsA * Tt + tkA * 16;
    const int k0b = nsB * Tt + tkB * 16;

    v16bf ka = load_frag(Kp + (size_t)(k0a + cl) * HD, 0, lane);
    v16bf kb = load_frag(Kp + (size_t)(k0b + cl) * HD, 0, lane);
    v8f z = {};
    v8f Sa = __builtin_amdgcn_wmma_f32_16x16x32_bf16(false, qf, false, ka, (short)0, z, false, false);
    v8f Sb = __builtin_amdgcn_wmma_f32_16x16x32_bf16(false, qf, false, kb, (short)0, z, false, false);

    if (tkA == tqt) {
#pragma unroll
      for (int g2 = 0; g2 < 8; ++g2)
        if (cl > rl + g2) Sa[g2] = -1e30f;
    }
    if (tkB == tqt) {
#pragma unroll
      for (int g2 = 0; g2 < 8; ++g2)
        if (cl > rl + g2) Sb[g2] = -1e30f;
    }

    // row-max over 16 columns (reduction stays within each 16-lane half)
    float mn[8], sc[8];
#pragma unroll
    for (int g2 = 0; g2 < 8; ++g2) mn[g2] = fmaxf(Sa[g2], Sb[g2]);
#pragma unroll
    for (int off = 1; off < 16; off <<= 1)
#pragma unroll
      for (int g2 = 0; g2 < 8; ++g2) mn[g2] = fmaxf(mn[g2], __shfl_xor(mn[g2], off, 32));
#pragma unroll
    for (int g2 = 0; g2 < 8; ++g2) {
      mn[g2] = fmaxf(mn[g2], m[g2]);
      sc[g2] = __expf(m[g2] - mn[g2]);
      m[g2]  = mn[g2];
    }

    float Pa[8], Pb[8], rs[8];
#pragma unroll
    for (int g2 = 0; g2 < 8; ++g2) {
      Pa[g2] = __expf(Sa[g2] - mn[g2]);
      Pb[g2] = __expf(Sb[g2] - mn[g2]);
      rs[g2] = Pa[g2] + Pb[g2];
    }
#pragma unroll
    for (int off = 1; off < 16; off <<= 1)
#pragma unroll
      for (int g2 = 0; g2 < 8; ++g2) rs[g2] += __shfl_xor(rs[g2], off, 32);
#pragma unroll
    for (int g2 = 0; g2 < 8; ++g2) ssum[g2] = ssum[g2] * sc[g2] + rs[g2];

    // Re-distribute P (C-layout) -> A-fragment layout via wave-private LDS.
#pragma unroll
    for (int g2 = 0; g2 < 8; ++g2) {
      Pl[w][rl + g2][cl]      = f2bf(Pa[g2]);
      Pl[w][rl + g2][16 + cl] = f2bf(Pb[g2]);
    }
    asm volatile("s_wait_dscnt 0" ::: "memory"); // order LDS stores vs loads (wave-local)
    v16bf pf = load_frag(&Pl[w][cl][0], 0, lane);

    // V fragments: Vt rows are contiguous over key index.
    v16bf v0 = load_frag2(Vp + (size_t)cl * L + k0a,        Vp + (size_t)cl * L + k0b,        lane);
    v16bf v1 = load_frag2(Vp + (size_t)(16 + cl) * L + k0a, Vp + (size_t)(16 + cl) * L + k0b, lane);

#pragma unroll
    for (int g2 = 0; g2 < 8; ++g2) { O0[g2] *= sc[g2]; O1[g2] *= sc[g2]; }
    O0 = __builtin_amdgcn_wmma_f32_16x16x32_bf16(false, pf, false, v0, (short)0, O0, false, false);
    O1 = __builtin_amdgcn_wmma_f32_16x16x32_bf16(false, pf, false, v1, (short)0, O1, false, false);
    asm volatile("" ::: "memory");
  }

#pragma unroll
  for (int g2 = 0; g2 < 8; ++g2) {
    float inv = 1.f / ssum[g2];
    int gr = q0 + rl + g2;
    ctx[(size_t)gr * D + head * HD + cl]      = f2bf(O0[g2] * inv);
    ctx[(size_t)gr * D + head * HD + 16 + cl] = f2bf(O1[g2] * inv);
  }
}

// out = h + alpha * (ctx @ out_w^T + out_b)
__global__ __launch_bounds__(256) void k_outproj(const ushort_t* __restrict__ ctx_bf,
                                                 const ushort_t* __restrict__ ow_bf,
                                                 const float* __restrict__ ob,
                                                 const float* __restrict__ h,
                                                 const float* __restrict__ alphap,
                                                 float* __restrict__ out) {
  const int lane = threadIdx.x & 31;
  const int wave = (blockIdx.x << 3) + (threadIdx.x >> 5);
  const int mt = wave >> 4, nt = wave & 15;
  v8f acc = wave_gemm(ctx_bf, D, ow_bf, D, mt * 16, nt * 16, D, lane);
  const float alpha = alphap[0];
  const int col   = nt * 16 + (lane & 15);
  const int rbase = mt * 16 + ((lane >> 4) << 3);
#pragma unroll
  for (int g2 = 0; g2 < 8; ++g2) {
    size_t idx = (size_t)(rbase + g2) * D + col;
    out[idx] = h[idx] + alpha * (acc[g2] + ob[col]);
  }
}

// ---------------- launcher ----------------

extern "C" void kernel_launch(void* const* d_in, const int* in_sizes, int n_in,
                              void* d_out, int out_size, void* d_ws, size_t ws_size,
                              hipStream_t stream) {
  (void)in_sizes; (void)n_in; (void)out_size; (void)ws_size;
  const float* h     = (const float*)d_in[0];
  const float* ln_g  = (const float*)d_in[1];
  const float* ln_b  = (const float*)d_in[2];
  const float* wq    = (const float*)d_in[3];
  const float* bq    = (const float*)d_in[4];
  const float* sym   = (const float*)d_in[5];
  const float* in_w  = (const float*)d_in[6];
  const float* in_b  = (const float*)d_in[7];
  const float* ow    = (const float*)d_in[8];
  const float* ob    = (const float*)d_in[9];
  const float* alpha = (const float*)d_in[10];
  float* out = (float*)d_out;

  char* p = (char*)d_ws;
  auto take = [&](size_t bytes) {
    char* r = p;
    p += (bytes + 255) & ~(size_t)255;
    return r;
  };
  ushort_t* x_bf   = (ushort_t*)take((size_t)L * D * 2);
  ushort_t* kv_bf  = (ushort_t*)take((size_t)L * D * 2);
  ushort_t* qf_bf  = (ushort_t*)take((size_t)L * D * 2);
  ushort_t* ctx_bf = (ushort_t*)take((size_t)L * D * 2);
  ushort_t* Qh     = (ushort_t*)take((size_t)Hh * L * HD * 2);
  ushort_t* Kh     = (ushort_t*)take((size_t)Hh * L * HD * 2);
  ushort_t* Vt     = (ushort_t*)take((size_t)Hh * HD * L * 2);
  ushort_t* wq_bf  = (ushort_t*)take((size_t)D * D * 2);
  ushort_t* inw_bf = (ushort_t*)take((size_t)3 * D * D * 2);
  ushort_t* ow_bf  = (ushort_t*)take((size_t)D * D * 2);

  // weight conversions
  k_convert_bf16<<<(D * D + 255) / 256, 256, 0, stream>>>(wq, wq_bf, D * D);
  k_convert_bf16<<<(3 * D * D + 255) / 256, 256, 0, stream>>>(in_w, inw_bf, 3 * D * D);
  k_convert_bf16<<<(D * D + 255) / 256, 256, 0, stream>>>(ow, ow_bf, D * D);

  // LayerNorm + bf16 copies
  k_ln<<<L, 256, 0, stream>>>(h, ln_g, ln_b, x_bf, kv_bf);

  // q projection (+ sym emb)
  k_qproj<<<(L / 16) * (D / 16) / 8, 256, 0, stream>>>(x_bf, wq_bf, bq, sym, qf_bf);

  // Q/K/V projections
  dim3 gqkv((L / 16) * (D / 16) / 8, 3);
  k_qkv<<<gqkv, 256, 0, stream>>>(qf_bf, kv_bf, inw_bf, in_b, Qh, Kh, Vt);

  // flash attention: 8 heads * 256 query tiles = 2048 waves
  k_attn<<<Hh * (L / 16) / 8, 256, 0, stream>>>(Qh, Kh, Vt, ctx_bf);

  // output projection + residual
  k_outproj<<<(L / 16) * (D / 16) / 8, 256, 0, stream>>>(ctx_bf, ow_bf, ob, h, alpha, out);
}
